// SelfAttention_66374424592529
// MI455X (gfx1250) — matrix operand order repaired and checked by reading, not compiled, not run on hardware
//
#include <hip/hip_runtime.h>
#include <hip/hip_bf16.h>

typedef __attribute__((ext_vector_type(16))) _Float16 v16h;
typedef __attribute__((ext_vector_type(8)))  _Float16 v8h;
typedef __attribute__((ext_vector_type(8)))  float    v8f;

#define BB   8
#define LL   2048
#define DH   256      // model dim of this attention block
#define HD   64       // head dim
#define NH   4
#define ROWS (BB*LL)  // 16384
#define DOUT 1024

#if defined(__AMDGCN__) && __has_builtin(__builtin_amdgcn_global_load_async_to_lds_b128)
#define USE_ASYNC_LDS 1
// The builtin takes pointers to a 16-byte gcc-vector of int (per hipcc's
// diagnostic: "__attribute__((__vector_size__(4 * sizeof(int)))) int *").
typedef int async_b128_t __attribute__((vector_size(16)));
#else
#define USE_ASYNC_LDS 0
#endif

// ---------------------------------------------------------------------------
// WMMA helpers (CDNA5 16x16x32 f16 -> f32). Layouts per ISA 7.12.2:
//  A (16x32, f16): lane l (m=l&15, hs=l>>4) holds K = {8*hs+i}, {8*hs+16+i}
//                  -> two contiguous 8-half runs (2x ds_load_b128).
//  B (32x16, f16): lane l (n=l&15, hs=l>>4) holds K = {8*hs+i}, {8*hs+16+i}
//                  -> two contiguous 8-half runs when B is stored [N][K].
//  C/D (16x16,f32): element r <-> row r + 8*hs, col l&15.
// ---------------------------------------------------------------------------
__device__ __forceinline__ v8f wmma_f16(v16h a, v16h b, v8f c) {
    return __builtin_amdgcn_wmma_f32_16x16x32_f16(false, a, false, b,
                                                  (short)0, c, false, false);
}

__device__ __forceinline__ v16h load_a_frag(const _Float16* base, int row_stride,
                                            int k0, int lane) {
    int m = lane & 15, hs = lane >> 4;
    const _Float16* p = base + m * row_stride + k0 + hs * 8;
    v8h lo = *(const v8h*)p;
    v8h hi = *(const v8h*)(p + 16);
    v16h a;
#pragma unroll
    for (int i = 0; i < 8; ++i) { a[i] = lo[i]; a[i + 8] = hi[i]; }
    return a;
}

// base points at a [16][row_stride] tile holding B transposed: base[n][kc]
__device__ __forceinline__ v16h load_b_frag(const _Float16* base, int row_stride,
                                            int k0, int lane) {
    int n = lane & 15, hs = lane >> 4;
    const _Float16* p = base + n * row_stride + k0 + hs * 8;
    v8h lo = *(const v8h*)p;
    v8h hi = *(const v8h*)(p + 16);
    v16h b;
#pragma unroll
    for (int i = 0; i < 8; ++i) { b[i] = lo[i]; b[i + 8] = hi[i]; }
    return b;
}

// Contiguous tile staging: global -> LDS, 16B per lane per step. Uses the
// CDNA5 async-to-LDS path (ASYNCcnt) when the builtin exists.
__device__ __forceinline__ void stage_tile(_Float16* lds, const _Float16* g,
                                           int n_uint4, int tid, int nthreads) {
    uint4* lp = (uint4*)lds;
    const uint4* gp = (const uint4*)g;
    for (int t = tid; t < n_uint4; t += nthreads) {
#if USE_ASYNC_LDS
        __builtin_amdgcn_global_load_async_to_lds_b128(
            (__attribute__((address_space(1))) async_b128_t*)(gp + t),
            (__attribute__((address_space(3))) async_b128_t*)(lp + t), 0, 0);
#else
        lp[t] = gp[t];
#endif
    }
}

__device__ __forceinline__ void wait_async_lds() {
#if USE_ASYNC_LDS
#if __has_builtin(__builtin_amdgcn_s_wait_asynccnt)
    __builtin_amdgcn_s_wait_asynccnt(0);
#else
    asm volatile("s_wait_asynccnt 0x0" ::: "memory");
#endif
#endif
}

__device__ __forceinline__ void wait_ds() {
    asm volatile("s_wait_dscnt 0x0" ::: "memory");
}

// ---------------------------------------------------------------------------
// Conversion kernels
// ---------------------------------------------------------------------------
__global__ void k_f32_to_f16(const float* __restrict__ in, _Float16* __restrict__ out, int n) {
    int i = blockIdx.x * blockDim.x + threadIdx.x;
    if (i < n) out[i] = (_Float16)in[i];
}

// in[K][N] (fp32, [in_dim][out_dim]) -> out[N][K] (f16, transposed)
__global__ void k_transpose_f16(const float* __restrict__ in, _Float16* __restrict__ out,
                                int K, int N) {
    int i = blockIdx.x * blockDim.x + threadIdx.x;
    if (i < K * N) {
        int k = i / N, n = i - k * N;
        out[n * K + k] = (_Float16)in[i];
    }
}

// ---------------------------------------------------------------------------
// QKV projection: 4-wave workgroup computes a 64x64 tile of Q/K/V
// (blockIdx.z picks which). Each wave: 16 rows x 64 cols, A-fragment reused
// across 4 WMMAs. Output head-split [B*H][L][HD] f16; Q pre-scaled by 1/16.
// ---------------------------------------------------------------------------
__global__ void k_qkv(const _Float16* __restrict__ xh,
                      const _Float16* __restrict__ wq_t,
                      const _Float16* __restrict__ wk_t,
                      const _Float16* __restrict__ wv_t,
                      const float* __restrict__ bq,
                      const float* __restrict__ bk,
                      const float* __restrict__ bv,
                      _Float16* __restrict__ Qo,
                      _Float16* __restrict__ Ko,
                      _Float16* __restrict__ Vo) {
    __shared__ __align__(32) _Float16 ldsX[64 * DH];  // 32KB
    __shared__ __align__(32) _Float16 ldsW[64 * DH];  // 32KB
    int tid  = threadIdx.x;
    int lane = tid & 31;
    int wv_i = tid >> 5;                 // wave index 0..3
    int row0 = blockIdx.x * 64;
    int col0 = blockIdx.y * 64;
    int mat  = blockIdx.z;
    const _Float16* wt   = (mat == 0) ? wq_t : (mat == 1) ? wk_t : wv_t;
    const float*    bias = (mat == 0) ? bq  : (mat == 1) ? bk  : bv;
    _Float16*       outp = (mat == 0) ? Qo  : (mat == 1) ? Ko  : Vo;

    stage_tile(ldsX, xh + row0 * DH, 2048, tid, 128);  // 64x256 halves
    stage_tile(ldsW, wt + col0 * DH, 2048, tid, 128);  // 64 Wt rows x 256
    wait_async_lds();
    wait_ds();
    __syncthreads();

    v8f acc[4] = {};
    const _Float16* aBase = ldsX + (wv_i * 16) * DH;
#pragma unroll
    for (int k0 = 0; k0 < DH; k0 += 32) {
        v16h a = load_a_frag(aBase, DH, k0, lane);
#pragma unroll
        for (int c = 0; c < 4; ++c)
            acc[c] = wmma_f16(a, load_b_frag(ldsW + (16 * c) * DH, DH, k0, lane), acc[c]);
    }

    int n = lane & 15, hs = lane >> 4;
    int head = blockIdx.y;               // 64-wide col block == one head
    float scl = (mat == 0) ? 0.0625f : 1.0f;  // 1/sqrt(256) folded into Q
#pragma unroll
    for (int c = 0; c < 4; ++c) {
        int col = col0 + 16 * c + n;
        float bval = bias[col];
        int dh = 16 * c + n;
#pragma unroll
        for (int r = 0; r < 8; ++r) {
            int row  = row0 + wv_i * 16 + r + 8 * hs;
            int bidx = row >> 11;        // /LL
            int lloc = row & (LL - 1);
            float v = (acc[c][r] + bval) * scl;
            outp[(((bidx * NH + head) * LL) + lloc) * HD + dh] = (_Float16)v;
        }
    }
}

// ---------------------------------------------------------------------------
// Flash attention: one wave per (bh, 16-query tile), 64 keys per step.
// ---------------------------------------------------------------------------
__global__ void k_attn(const _Float16* __restrict__ Q,
                       const _Float16* __restrict__ K,
                       const _Float16* __restrict__ V,
                       _Float16* __restrict__ ctx) {
    __shared__ __align__(32) _Float16 ldsQ[16 * HD];   // [q][hd]     2KB
    __shared__ __align__(32) _Float16 ldsK[64 * HD];   // [key][hd]   8KB
    __shared__ __align__(32) _Float16 ldsVt[HD * 64];  // [hd][key]   8KB
    __shared__ __align__(32) _Float16 ldsP[16 * 64];   // [q][key]    2KB
    int lane = threadIdx.x;
    int q0 = blockIdx.x * 16;
    int bh = blockIdx.y;  // b*NH + h
    const _Float16* Qb = Q + (bh * LL + q0) * HD;
    const _Float16* Kb = K + bh * LL * HD;
    const _Float16* Vb = V + bh * LL * HD;

    stage_tile(ldsQ, Qb, 128, lane, 32);
    wait_async_lds();
    wait_ds();
    __syncthreads();
    v16h qa0 = load_a_frag(ldsQ, HD, 0, lane);
    v16h qa1 = load_a_frag(ldsQ, HD, 32, lane);

    v8f acc[4] = {};
    float m_i[8], l_i[8];
#pragma unroll
    for (int r = 0; r < 8; ++r) { m_i[r] = -1e30f; l_i[r] = 0.0f; }

    int n = lane & 15, hs = lane >> 4;

    for (int j0 = 0; j0 < LL; j0 += 64) {
        wait_ds();          // previous iteration's LDS reads complete
        __syncthreads();    // before async writes overwrite the tiles
        // K tile [64][64] via async-to-LDS (ASYNCcnt path).
        stage_tile(ldsK, Kb + j0 * HD, 512, lane, 32);
        // V tile transposed into [hd][key] (ds scatter).
        {
            const unsigned int* g32 = (const unsigned int*)(Vb + j0 * HD);
            for (int t = lane; t < 2048; t += 32) {
                unsigned int uval = g32[t];
                int key = t >> 5;        // 32 dwords (64 halves) per key
                int dp  = t & 31;
                ldsVt[(2 * dp) * 64 + key] =
                    __builtin_bit_cast(_Float16, (unsigned short)(uval & 0xffffu));
                ldsVt[(2 * dp + 1) * 64 + key] =
                    __builtin_bit_cast(_Float16, (unsigned short)(uval >> 16));
            }
        }
        if (j0 + 64 < LL) {  // prefetch next tiles (global_prefetch_b8)
            __builtin_prefetch(Kb + (j0 + 64) * HD, 0, 1);
            __builtin_prefetch(Vb + (j0 + 64) * HD, 0, 1);
        }
        wait_async_lds();
        wait_ds();
        __syncthreads();

        // scores = Q*K^T: 4 N-tiles of 16 keys, hd=64 -> 2 k-steps each.
        v8f s[4];
#pragma unroll
        for (int t2 = 0; t2 < 4; ++t2) {
            v8f z = {};
            z = wmma_f16(qa0, load_b_frag(ldsK + (16 * t2) * HD, HD,  0, lane), z);
            z = wmma_f16(qa1, load_b_frag(ldsK + (16 * t2) * HD, HD, 32, lane), z);
            s[t2] = z;
        }

        // Online softmax per row (element r <-> row r + 8*hs).
#pragma unroll
        for (int r = 0; r < 8; ++r) {
            float rv = fmaxf(fmaxf(s[0][r], s[1][r]), fmaxf(s[2][r], s[3][r]));
#pragma unroll
            for (int mm = 1; mm <= 8; mm <<= 1) rv = fmaxf(rv, __shfl_xor(rv, mm, 32));
            float m_new = fmaxf(m_i[r], rv);
            float p[4], rs = 0.0f;
#pragma unroll
            for (int t2 = 0; t2 < 4; ++t2) { p[t2] = __expf(s[t2][r] - m_new); rs += p[t2]; }
#pragma unroll
            for (int mm = 1; mm <= 8; mm <<= 1) rs += __shfl_xor(rs, mm, 32);
            float resc = __expf(m_i[r] - m_new);
            l_i[r] = l_i[r] * resc + rs;
            m_i[r] = m_new;
#pragma unroll
            for (int c = 0; c < 4; ++c) acc[c][r] *= resc;
            int rowm = r + 8 * hs;
#pragma unroll
            for (int t2 = 0; t2 < 4; ++t2)
                ldsP[rowm * 64 + 16 * t2 + n] = (_Float16)p[t2];
        }
        wait_ds();
        __syncthreads();

        // ctx += P(16x64) * V(64x64): 2 k-steps x 4 N-tiles of dims.
        v16h pa0 = load_a_frag(ldsP, 64, 0, lane);
        v16h pa1 = load_a_frag(ldsP, 64, 32, lane);
#pragma unroll
        for (int c = 0; c < 4; ++c) {
            acc[c] = wmma_f16(pa0, load_b_frag(ldsVt + (16 * c) * 64, 64,  0, lane), acc[c]);
            acc[c] = wmma_f16(pa1, load_b_frag(ldsVt + (16 * c) * 64, 64, 32, lane), acc[c]);
        }
    }

    // Normalize and write ctx back to [B][L][H*64] layout (f16).
    int b = bh >> 2, h = bh & (NH - 1);
#pragma unroll
    for (int r = 0; r < 8; ++r) {
        float inv = 1.0f / l_i[r];
        int lloc = q0 + r + 8 * hs;
        _Float16* op = ctx + (b * LL + lloc) * DH + h * HD;
#pragma unroll
        for (int c = 0; c < 4; ++c)
            op[16 * c + n] = (_Float16)(acc[c][r] * inv);
    }
}

// ---------------------------------------------------------------------------
// Output projection: ctx[16384x256] @ Wo[256x1024] + bo -> fp32 out.
// 4-wave workgroup computes a 64x64 fp32 tile.
// ---------------------------------------------------------------------------
__global__ void k_oproj(const _Float16* __restrict__ ctx,
                        const _Float16* __restrict__ wo_t,
                        const float* __restrict__ bo,
                        float* __restrict__ out) {
    __shared__ __align__(32) _Float16 ldsX[64 * DH];
    __shared__ __align__(32) _Float16 ldsW[64 * DH];
    int tid  = threadIdx.x;
    int lane = tid & 31;
    int wv_i = tid >> 5;
    int row0 = blockIdx.x * 64;
    int col0 = blockIdx.y * 64;

    stage_tile(ldsX, ctx  + row0 * DH, 2048, tid, 128);
    stage_tile(ldsW, wo_t + col0 * DH, 2048, tid, 128);
    wait_async_lds();
    wait_ds();
    __syncthreads();

    v8f acc[4] = {};
    const _Float16* aBase = ldsX + (wv_i * 16) * DH;
#pragma unroll
    for (int k0 = 0; k0 < DH; k0 += 32) {
        v16h a = load_a_frag(aBase, DH, k0, lane);
#pragma unroll
        for (int c = 0; c < 4; ++c)
            acc[c] = wmma_f16(a, load_b_frag(ldsW + (16 * c) * DH, DH, k0, lane), acc[c]);
    }
    int n = lane & 15, hs = lane >> 4;
#pragma unroll
    for (int c = 0; c < 4; ++c) {
        int col = col0 + 16 * c + n;
        float bv = bo[col];
#pragma unroll
        for (int r = 0; r < 8; ++r) {
            int row = row0 + wv_i * 16 + r + 8 * hs;
            out[row * DOUT + col] = acc[c][r] + bv;
        }
    }
}

// ---------------------------------------------------------------------------
extern "C" void kernel_launch(void* const* d_in, const int* in_sizes, int n_in,
                              void* d_out, int out_size, void* d_ws, size_t ws_size,
                              hipStream_t stream) {
    const float* x  = (const float*)d_in[0];
    const float* Wq = (const float*)d_in[1];
    const float* bq = (const float*)d_in[2];
    const float* Wk = (const float*)d_in[3];
    const float* bk = (const float*)d_in[4];
    const float* Wv = (const float*)d_in[5];
    const float* bv = (const float*)d_in[6];
    const float* Wo = (const float*)d_in[7];
    const float* bo = (const float*)d_in[8];
    float* out = (float*)d_out;

    // Workspace partition (f16 halves), ~43 MB total.
    _Float16* ws   = (_Float16*)d_ws;
    _Float16* xh   = ws;                         // 16384*256   = 4194304
    _Float16* wqt  = xh  + (size_t)ROWS * DH;    // 256*256     = 65536
    _Float16* wkt  = wqt + DH * DH;
    _Float16* wvt  = wkt + DH * DH;
    _Float16* wot  = wvt + DH * DH;              // 1024*256    = 262144
    _Float16* Qw   = wot + (size_t)DOUT * DH;    // 32*2048*64  = 4194304
    _Float16* Kw   = Qw  + (size_t)BB * NH * LL * HD;
    _Float16* Vw   = Kw  + (size_t)BB * NH * LL * HD;
    _Float16* ctxh = Vw  + (size_t)BB * NH * LL * HD;

    // 1) fp32 -> f16 converts (+ weight transposes so B-frags are contiguous)
    {
        int n = ROWS * DH;
        k_f32_to_f16<<<(n + 255) / 256, 256, 0, stream>>>(x, xh, n);
    }
    k_transpose_f16<<<(DH * DH + 255) / 256, 256, 0, stream>>>(Wq, wqt, DH, DH);
    k_transpose_f16<<<(DH * DH + 255) / 256, 256, 0, stream>>>(Wk, wkt, DH, DH);
    k_transpose_f16<<<(DH * DH + 255) / 256, 256, 0, stream>>>(Wv, wvt, DH, DH);
    k_transpose_f16<<<(DH * DOUT + 255) / 256, 256, 0, stream>>>(Wo, wot, DH, DOUT);

    // 2) QKV projections (head-split f16, Q pre-scaled)
    k_qkv<<<dim3(ROWS / 64, DH / 64, 3), 128, 0, stream>>>(
        xh, wqt, wkt, wvt, bq, bk, bv, Qw, Kw, Vw);

    // 3) flash attention
    k_attn<<<dim3(LL / 16, BB * NH), 32, 0, stream>>>(Qw, Kw, Vw, ctxh);

    // 4) output projection
    k_oproj<<<dim3(ROWS / 64, DOUT / 64), 128, 0, stream>>>(ctxh, wot, bo, out);
}
